// Precomputation_67482526155485
// MI455X (gfx1250) — matrix operand order repaired and checked by reading, not compiled, not run on hardware
//
#include <hip/hip_runtime.h>

typedef __attribute__((ext_vector_type(16))) _Float16 v16h;
typedef __attribute__((ext_vector_type(8)))  _Float16 v8h;
typedef __attribute__((ext_vector_type(8)))  float    v8f;

#define N_SNAP 256
#define N_ATOMS 64
#define AEV_DIM 384
#define N_ENS 8
#define OUT_DIM 96
#define MAX_DIM 160
#define STRIDE1 160   // padded h1 (multiple of 32)
#define STRIDE2 128   // padded h2 (multiple of 32)
#define NMAX 160      // max layer output width
#define CHUNK (NMAX * 32)   // halves per WT chunk buffer

// Precomputed f16 transposed-weight pool offsets in d_ws (units: halves).
#define WS_L2_BASE 1720320
#define WS_L3_BASE 2314240
#define WS_W_HALVES 2707456ull
#define AEV_HALVES (256ull * 64 * 384)   // 6291456

struct AniW {
    const float* W1[4]; const float* B1[4];
    const float* W2[4]; const float* B2[4];
    const float* W3[4]; const float* B3[4];
};

template <typename T>
__device__ __forceinline__ T sel4(int s, T a, T b, T c, T d) {
    return s == 0 ? a : (s == 1 ? b : (s == 2 ? c : d));
}

__device__ __forceinline__ float celu01(float x) {
    return x > 0.f ? x : 0.1f * (__expf(x * 10.f) - 1.f);
}

__device__ __forceinline__ v8f wmma_f16(v16h a, v16h b, v8f c) {
    return __builtin_amdgcn_wmma_f32_16x16x32_f16(false, a, false, b, (short)0, c,
                                                  false, false);
}

// A fragment (16x32 f16) from a global f32 row. p = row base + kt*32.
__device__ __forceinline__ v16h load_a_f32(const float* __restrict__ p, int g) {
    const float* p0 = p + g * 8;
    const float* p1 = p + 16 + g * 8;
    v16h a;
#pragma unroll
    for (int j = 0; j < 8; ++j) a[j] = (_Float16)p0[j];
#pragma unroll
    for (int j = 0; j < 8; ++j) a[8 + j] = (_Float16)p1[j];
    return a;
}

// A fragment from f16 memory (LDS or pre-converted global). p = row base + kt*32.
__device__ __forceinline__ v16h load_a_h16(const _Float16* p, int g) {
    v8h lo = *(const v8h*)(p + g * 8);
    v8h hi = *(const v8h*)(p + 16 + g * 8);
    v16h a;
#pragma unroll
    for (int j = 0; j < 8; ++j) { a[j] = lo[j]; a[8 + j] = hi[j]; }
    return a;
}

// B fragment (32x16 f16) from LDS transposed chunk WT[n][kk] (row = 32 halves).
__device__ __forceinline__ v16h load_b_wt(const _Float16* WT, int n, int g) {
    const _Float16* p = WT + n * 32 + g * 8;
    v8h lo = *(const v8h*)(p);
    v8h hi = *(const v8h*)(p + 16);
    v16h b;
#pragma unroll
    for (int j = 0; j < 8; ++j) { b[j] = lo[j]; b[8 + j] = hi[j]; }
    return b;
}

// ---- Fallback staging from f32 global weights (convert + transpose in-kernel) ----
__device__ __forceinline__ void stage_load(const float* __restrict__ W, int N, int K,
                                           int kbase, int tid, float4* wr) {
#pragma unroll
    for (int it = 0; it < 10; ++it) {
        const int i  = tid + it * 128;
        const int kk = i / 40;
        const int q  = i - kk * 40;
        const int n0 = q * 4;
        const int k  = kbase + kk;
        float4 w = make_float4(0.f, 0.f, 0.f, 0.f);
        if (n0 < N && k < K) w = *(const float4*)(W + (size_t)k * N + n0);
        wr[it] = w;
    }
}

__device__ __forceinline__ void stage_store(_Float16* __restrict__ WT, int N, int tid,
                                            const float4* wr) {
#pragma unroll
    for (int it = 0; it < 10; ++it) {
        const int i  = tid + it * 128;
        const int kk = i / 40;
        const int q  = i - kk * 40;
        const int n0 = q * 4;
        if (n0 < N) {
            WT[(n0 + 0) * 32 + kk] = (_Float16)wr[it].x;
            WT[(n0 + 1) * 32 + kk] = (_Float16)wr[it].y;
            WT[(n0 + 2) * 32 + kk] = (_Float16)wr[it].z;
            WT[(n0 + 3) * 32 + kk] = (_Float16)wr[it].w;
        }
    }
}

// ---- PREP staging: chunk already f16/transposed/padded -> straight b128 copy ----
__device__ __forceinline__ void stage_load_prep(const _Float16* __restrict__ src,
                                                int nv8, int tid, v8h* wr8) {
    const v8h* s = (const v8h*)src;
#pragma unroll
    for (int it = 0; it < 5; ++it) {
        const int i = tid + it * 128;
        if (i < nv8) wr8[it] = s[i];
    }
}

__device__ __forceinline__ void stage_store_prep(_Float16* __restrict__ dst,
                                                 int nv8, int tid, const v8h* wr8) {
    v8h* d = (v8h*)dst;
#pragma unroll
    for (int it = 0; it < 5; ++it) {
        const int i = tid + it * 128;
        if (i < nv8) d[i] = wr8[it];
    }
}

// ---- Prep kernels ----
__global__ __launch_bounds__(128) void ani_prep_wt(AniW P, _Float16* __restrict__ WP,
                                                   int layer) {
    const int kt = blockIdx.x, e = blockIdx.y, sp = blockIdx.z, tid = threadIdx.x;
    const int h1 = sel4(sp, 160, 144, 128, 128);
    const int h2 = sel4(sp, 128, 112, 112, 112);
    int h, K; const float* W; size_t dst;
    if (layer == 0) {
        h = h1; K = AEV_DIM;
        W = P.W1[sp] + (size_t)e * AEV_DIM * h1;
        dst = (size_t)sel4(sp, 0, 491520, 933888, 1327104) + (size_t)e * 12 * h1 * 32;
    } else if (layer == 1) {
        h = h2; K = h1;
        W = P.W2[sp] + (size_t)e * h1 * h2;
        dst = WS_L2_BASE + (size_t)sel4(sp, 0, 163840, 307200, 450560)
            + (size_t)e * 5 * h2 * 32;
    } else {
        h = OUT_DIM; K = h2;
        W = P.W3[sp] + (size_t)e * h2 * OUT_DIM;
        dst = WS_L3_BASE + (size_t)sp * 98304 + (size_t)e * 12288;
    }
    _Float16* D = WP + dst + (size_t)kt * h * 32;
    for (int i = tid; i < 32 * 40; i += 128) {
        const int kk = i / 40;
        const int q  = i - kk * 40;
        const int n0 = q * 4;
        if (n0 < h) {
            const int k = kt * 32 + kk;
            float4 w = make_float4(0.f, 0.f, 0.f, 0.f);
            if (k < K) w = *(const float4*)(W + (size_t)k * h + n0);
            D[(n0 + 0) * 32 + kk] = (_Float16)w.x;
            D[(n0 + 1) * 32 + kk] = (_Float16)w.y;
            D[(n0 + 2) * 32 + kk] = (_Float16)w.z;
            D[(n0 + 3) * 32 + kk] = (_Float16)w.w;
        }
    }
}

__global__ __launch_bounds__(256) void ani_prep_aev(const float* __restrict__ aev,
                                                    _Float16* __restrict__ AV) {
    const size_t i = ((size_t)blockIdx.x * 256 + threadIdx.x) * 8;
    float4 a = *(const float4*)(aev + i);
    float4 b = *(const float4*)(aev + i + 4);
    v8h h;
    h[0] = (_Float16)a.x; h[1] = (_Float16)a.y; h[2] = (_Float16)a.z; h[3] = (_Float16)a.w;
    h[4] = (_Float16)b.x; h[5] = (_Float16)b.y; h[6] = (_Float16)b.z; h[7] = (_Float16)b.w;
    *(v8h*)(AV + i) = h;
}

// MODE: 0 = no ws; 1 = weights prepped; 2 = weights + aev-f16 prepped
template <int MODE>
__global__ __launch_bounds__(128) void ani_mlp_wmma(
    const float* __restrict__ aev, const int* __restrict__ species,
    float* __restrict__ out, AniW P, const _Float16* __restrict__ WP,
    const _Float16* __restrict__ AV)
{
    __shared__ __align__(16) _Float16 smem[64 * STRIDE1 + 64 * STRIDE2 + 2 * CHUNK];
    _Float16* H1 = smem;
    _Float16* H2 = smem + 64 * STRIDE1;
    _Float16* WT = smem + 64 * STRIDE1 + 64 * STRIDE2;

    const int tid  = threadIdx.x;
    const int lane = tid & 31;
    const int wid  = tid >> 5;
    const int g    = lane >> 4;
    const int ln   = lane & 15;
    const int mbase = blockIdx.x * 64;
    const int atom  = blockIdx.y;
    const int e     = blockIdx.z;
    const int m0    = wid * 16;

    const int sp = species[atom];
    const int h1 = sel4(sp, 160, 144, 128, 128);
    const int h2 = sel4(sp, 128, 112, 112, 112);
    const int nT1 = h1 >> 4;
    const int nT2 = h2 >> 4;

    const float* W1 = sel4(sp, P.W1[0], P.W1[1], P.W1[2], P.W1[3]) + (size_t)e * AEV_DIM * h1;
    const float* B1 = sel4(sp, P.B1[0], P.B1[1], P.B1[2], P.B1[3]) + (size_t)e * h1;
    const float* W2 = sel4(sp, P.W2[0], P.W2[1], P.W2[2], P.W2[3]) + (size_t)e * h1 * h2;
    const float* B2 = sel4(sp, P.B2[0], P.B2[1], P.B2[2], P.B2[3]) + (size_t)e * h2;
    const float* W3 = sel4(sp, P.W3[0], P.W3[1], P.W3[2], P.W3[3]) + (size_t)e * h2 * OUT_DIM;
    const float* B3 = sel4(sp, P.B3[0], P.B3[1], P.B3[2], P.B3[3]) + (size_t)e * OUT_DIM;

    const _Float16* WP1 = nullptr;
    const _Float16* WP2 = nullptr;
    const _Float16* WP3 = nullptr;
    if constexpr (MODE >= 1) {
        WP1 = WP + (size_t)sel4(sp, 0, 491520, 933888, 1327104) + (size_t)e * 12 * h1 * 32;
        WP2 = WP + WS_L2_BASE + (size_t)sel4(sp, 0, 163840, 307200, 450560)
            + (size_t)e * 5 * h2 * 32;
        WP3 = WP + WS_L3_BASE + (size_t)sp * 98304 + (size_t)e * 12288;
    }

    // Zero H1+H2 so K dims not divisible by 32 (h1=144, h2=112) read zeros past edge.
    {
        uint4* Z = (uint4*)smem;
        const int n16 = (64 * STRIDE1 + 64 * STRIDE2) * 2 / 16;
        for (int i = tid; i < n16; i += 128) Z[i] = make_uint4(0u, 0u, 0u, 0u);
    }
    __syncthreads();

    float4 wr[10];   // fallback staging regs
    v8h   wr8[5];    // prep staging regs

    // ---------------- Layer 1: (64x384) x (384xh1) ----------------
    {
        const int KT = AEV_DIM / 32;  // 12
        v8f acc[10] = {};
        const size_t rowoff = ((size_t)((mbase + m0 + ln) * N_ATOMS + atom)) * AEV_DIM;
        const float*    arow32 = aev + rowoff;
        const _Float16* arow16 = (MODE == 2) ? (AV + rowoff) : nullptr;
        if constexpr (MODE >= 1) {
            stage_load_prep(WP1, h1 * 4, tid, wr8);
            stage_store_prep(WT, h1 * 4, tid, wr8);
        } else {
            stage_load(W1, h1, AEV_DIM, 0, tid, wr);
            stage_store(WT, h1, tid, wr);
        }
        __syncthreads();
        for (int kt = 0; kt < KT; ++kt) {
            _Float16* cur = WT + (kt & 1) * CHUNK;
            _Float16* nxt = WT + ((kt + 1) & 1) * CHUNK;
            const bool more = (kt + 1 < KT);
            if (more) {
                if constexpr (MODE >= 1) {
                    stage_load_prep(WP1 + (size_t)(kt + 1) * h1 * 32, h1 * 4, tid, wr8);
                    if (kt + 2 < KT)
                        __builtin_prefetch(WP1 + (size_t)(kt + 2) * h1 * 32 + tid * 40);
                } else {
                    stage_load(W1, h1, AEV_DIM, (kt + 1) * 32, tid, wr);
                    if (kt + 2 < KT)
                        __builtin_prefetch(W1 + (size_t)(kt + 2) * 32 * h1 + tid * 16);
                }
            }
            v16h a;
            if constexpr (MODE == 2) a = load_a_h16(arow16 + kt * 32, g);
            else                     a = load_a_f32(arow32 + kt * 32, g);
            v16h bc = load_b_wt(cur, ln, g);
#pragma unroll
            for (int nt = 0; nt < 10; ++nt) {
                if (nt < nT1) {   // wave-uniform guards: EXEC stays full for WMMA
                    v16h bn;
                    if (nt + 1 < nT1) bn = load_b_wt(cur, (nt + 1) * 16 + ln, g);
                    acc[nt] = wmma_f16(a, bc, acc[nt]);
                    if (nt + 1 < nT1) bc = bn;
                }
            }
            if (more) {
                if constexpr (MODE >= 1) stage_store_prep(nxt, h1 * 4, tid, wr8);
                else                     stage_store(nxt, h1, tid, wr);
            }
            __syncthreads();
        }
#pragma unroll
        for (int nt = 0; nt < 10; ++nt) {
            if (nt < nT1) {
                const int n = nt * 16 + ln;
                const float bias = B1[n];
#pragma unroll
                for (int r = 0; r < 8; ++r) {
                    const int m = m0 + r + 8 * g;
                    H1[m * STRIDE1 + n] = (_Float16)celu01(acc[nt][r] + bias);
                }
            }
        }
    }
    __syncthreads();

    // ---------------- Layer 2: (64xh1) x (h1xh2), K padded to 160 ----------------
    {
        const int KT = STRIDE1 / 32;  // 5
        v8f acc[8] = {};
        const _Float16* hrow = H1 + (m0 + ln) * STRIDE1;
        if constexpr (MODE >= 1) {
            stage_load_prep(WP2, h2 * 4, tid, wr8);
            stage_store_prep(WT, h2 * 4, tid, wr8);
        } else {
            stage_load(W2, h2, h1, 0, tid, wr);
            stage_store(WT, h2, tid, wr);
        }
        __syncthreads();
        for (int kt = 0; kt < KT; ++kt) {
            _Float16* cur = WT + (kt & 1) * CHUNK;
            _Float16* nxt = WT + ((kt + 1) & 1) * CHUNK;
            const bool more = (kt + 1 < KT);
            if (more) {
                if constexpr (MODE >= 1)
                    stage_load_prep(WP2 + (size_t)(kt + 1) * h2 * 32, h2 * 4, tid, wr8);
                else
                    stage_load(W2, h2, h1, (kt + 1) * 32, tid, wr);
            }
            v16h a = load_a_h16(hrow + kt * 32, g);
            v16h bc = load_b_wt(cur, ln, g);
#pragma unroll
            for (int nt = 0; nt < 8; ++nt) {
                if (nt < nT2) {
                    v16h bn;
                    if (nt + 1 < nT2) bn = load_b_wt(cur, (nt + 1) * 16 + ln, g);
                    acc[nt] = wmma_f16(a, bc, acc[nt]);
                    if (nt + 1 < nT2) bc = bn;
                }
            }
            if (more) {
                if constexpr (MODE >= 1) stage_store_prep(nxt, h2 * 4, tid, wr8);
                else                     stage_store(nxt, h2, tid, wr);
            }
            __syncthreads();
        }
#pragma unroll
        for (int nt = 0; nt < 8; ++nt) {
            if (nt < nT2) {
                const int n = nt * 16 + ln;
                const float bias = B2[n];
#pragma unroll
                for (int r = 0; r < 8; ++r) {
                    const int m = m0 + r + 8 * g;
                    H2[m * STRIDE2 + n] = (_Float16)celu01(acc[nt][r] + bias);
                }
            }
        }
    }
    __syncthreads();

    // ---------------- Layer 3: (64xh2) x (h2x96), K padded to 128 ----------------
    {
        const int KT = STRIDE2 / 32;  // 4
        v8f acc[6] = {};
        const _Float16* hrow = H2 + (m0 + ln) * STRIDE2;
        if constexpr (MODE >= 1) {
            stage_load_prep(WP3, OUT_DIM * 4, tid, wr8);
            stage_store_prep(WT, OUT_DIM * 4, tid, wr8);
        } else {
            stage_load(W3, OUT_DIM, h2, 0, tid, wr);
            stage_store(WT, OUT_DIM, tid, wr);
        }
        __syncthreads();
        for (int kt = 0; kt < KT; ++kt) {
            _Float16* cur = WT + (kt & 1) * CHUNK;
            _Float16* nxt = WT + ((kt + 1) & 1) * CHUNK;
            const bool more = (kt + 1 < KT);
            if (more) {
                if constexpr (MODE >= 1)
                    stage_load_prep(WP3 + (size_t)(kt + 1) * OUT_DIM * 32, OUT_DIM * 4,
                                    tid, wr8);
                else
                    stage_load(W3, OUT_DIM, h2, (kt + 1) * 32, tid, wr);
            }
            v16h a = load_a_h16(hrow + kt * 32, g);
            v16h bc = load_b_wt(cur, ln, g);
#pragma unroll
            for (int nt = 0; nt < 6; ++nt) {
                v16h bn;
                if (nt + 1 < 6) bn = load_b_wt(cur, (nt + 1) * 16 + ln, g);
                acc[nt] = wmma_f16(a, bc, acc[nt]);
                if (nt + 1 < 6) bc = bn;
            }
            if (more) {
                if constexpr (MODE >= 1) stage_store_prep(nxt, OUT_DIM * 4, tid, wr8);
                else                     stage_store(nxt, OUT_DIM, tid, wr);
            }
            __syncthreads();
        }
#pragma unroll
        for (int nt = 0; nt < 6; ++nt) {
            const int n = nt * 16 + ln;
            const float bias = B3[n];
#pragma unroll
            for (int r = 0; r < 8; ++r) {
                const int s = mbase + m0 + r + 8 * g;
                out[(((size_t)(s * N_ATOMS + atom)) * N_ENS + e) * MAX_DIM + n] =
                    celu01(acc[nt][r] + bias);
            }
        }
    }

    // ---------------- Zero-fill padding columns 96..159 ----------------
    {
        const float4 z = make_float4(0.f, 0.f, 0.f, 0.f);
        for (int i = tid; i < 64 * 16; i += 128) {
            const int row = i >> 4, c = i & 15;
            const int s = mbase + row;
            float4* p = (float4*)(out + (((size_t)(s * N_ATOMS + atom)) * N_ENS + e) * MAX_DIM
                                  + OUT_DIM) + c;
            *p = z;
        }
    }
}

extern "C" void kernel_launch(void* const* d_in, const int* in_sizes, int n_in,
                              void* d_out, int out_size, void* d_ws, size_t ws_size,
                              hipStream_t stream) {
    (void)in_sizes; (void)n_in; (void)out_size;
    const int*   species = (const int*)d_in[0];    // (256, 64) int32
    const float* aev     = (const float*)d_in[1];  // (256, 64, 384) f32

    AniW P;
    for (int s = 0; s < 4; ++s) {
        const int b = 2 + 6 * s;   // params flattened: (W1,b1,W2,b2,W3,b3) per species
        P.W1[s] = (const float*)d_in[b + 0];
        P.B1[s] = (const float*)d_in[b + 1];
        P.W2[s] = (const float*)d_in[b + 2];
        P.B2[s] = (const float*)d_in[b + 3];
        P.W3[s] = (const float*)d_in[b + 4];
        P.B3[s] = (const float*)d_in[b + 5];
    }

    dim3 grid(N_SNAP / 64, N_ATOMS, N_ENS);  // (4, 64, 8) = 2048 blocks
    dim3 block(128);                         // 4 wave32 waves

    const size_t ws_w    = WS_W_HALVES * sizeof(_Float16);                 // ~5.42 MB
    const size_t ws_full = (WS_W_HALVES + AEV_HALVES) * sizeof(_Float16);  // ~18.0 MB

    if (ws_size >= ws_full) {
        _Float16* WP = (_Float16*)d_ws;
        _Float16* AV = WP + WS_W_HALVES;
        ani_prep_wt<<<dim3(12, 8, 4), dim3(128), 0, stream>>>(P, WP, 0);
        ani_prep_wt<<<dim3(5,  8, 4), dim3(128), 0, stream>>>(P, WP, 1);
        ani_prep_wt<<<dim3(4,  8, 4), dim3(128), 0, stream>>>(P, WP, 2);
        ani_prep_aev<<<dim3((unsigned)(AEV_HALVES / (256 * 8))), dim3(256), 0, stream>>>(aev, AV);
        ani_mlp_wmma<2><<<grid, block, 0, stream>>>(aev, species, (float*)d_out, P, WP, AV);
    } else if (ws_size >= ws_w) {
        _Float16* WP = (_Float16*)d_ws;
        ani_prep_wt<<<dim3(12, 8, 4), dim3(128), 0, stream>>>(P, WP, 0);
        ani_prep_wt<<<dim3(5,  8, 4), dim3(128), 0, stream>>>(P, WP, 1);
        ani_prep_wt<<<dim3(4,  8, 4), dim3(128), 0, stream>>>(P, WP, 2);
        ani_mlp_wmma<1><<<grid, block, 0, stream>>>(aev, species, (float*)d_out, P, WP,
                                                    nullptr);
    } else {
        ani_mlp_wmma<0><<<grid, block, 0, stream>>>(aev, species, (float*)d_out, P,
                                                    nullptr, nullptr);
    }
}
